// DREAMCell_12154757448308
// MI455X (gfx1250) — matrix operand order repaired and checked by reading, not compiled, not run on hardware
//
#include <hip/hip_runtime.h>
#include <hip/hip_bf16.h>
#include <math.h>

// ---------------- problem constants ----------------
constexpr int kB = 64, kT = 64, kD = 1024, kH = 2048, kR = 64;
constexpr float kDT = 0.1f, kFORGET = 0.01f, kTARGET = 1.0f;
constexpr float kBETA = 0.1f, kBETA_S = 0.1f;
constexpr float kSLEEP = 0.01f, kSMIN = 0.3f;
constexpr float kSURP = 2.0f, kEPS = 1e-6f;
constexpr float kTWO_PI_E = 17.079468445347132f; // 2*pi*e
constexpr int kChunks = 8;                       // U-state split per batch row

typedef __attribute__((ext_vector_type(16))) _Float16 v16h;
typedef __attribute__((ext_vector_type(8)))  float    v8f;

union Frag16 { uint4 q[2]; v16h v; };

__device__ __forceinline__ v8f wmma_f16(const v16h& a, const v16h& b, const v8f& c) {
    return __builtin_amdgcn_wmma_f32_16x16x32_f16(
        /*neg_a=*/false, a, /*neg_b=*/false, b,
        /*c_mod=*/(short)0, c, /*reuse_a=*/false, /*reuse_b=*/false);
}

// ---------------- conversion / transpose ----------------
__global__ __launch_bounds__(256) void cvt_f16(const float* __restrict__ src,
                                               _Float16* __restrict__ dst, int n) {
    int i = blockIdx.x * 256 + threadIdx.x;
    if (i < n) dst[i] = (_Float16)src[i];
}

// src [rows, cols] -> dst [cols, rows] in f16
__global__ __launch_bounds__(256) void cvt_f16_T(const float* __restrict__ src,
                                                 _Float16* __restrict__ dst,
                                                 int rows, int cols) {
    int i = blockIdx.x * 256 + threadIdx.x;
    if (i >= rows * cols) return;
    int r = i / cols, c = i - r * cols;
    dst[(size_t)c * rows + r] = (_Float16)src[i];
}

// ---------------- generic WMMA f16 GEMM ----------------
// C[M,N] = A[M,K] * Bt[N,K]^T
// 128 threads = 4 waves; each wave computes a 16x32 output slab (two 16x16
// tiles sharing one A fragment -> 2 WMMAs per 6 B128 loads).
// Requires N % 32 == 0, K % 32 == 0, M % 16 == 0.
// epi 0: store fp32 to Cout (ldc)
// epi 1: e = Xres[m*ldx+n] - tanh(acc); store e to ErrF[m*N+n] fp32 and ErrH f16
#define EPI_STORE 0
#define EPI_TANH_ERR 1

__global__ __launch_bounds__(128)
void gemm_f16(const _Float16* __restrict__ A, int lda,
              const _Float16* __restrict__ Bt, int ldb,
              int M, int N, int K,
              float* __restrict__ Cout, int ldc, int epi,
              const float* __restrict__ Xres, int ldx,
              float* __restrict__ ErrF, _Float16* __restrict__ ErrH) {
    const int lane = threadIdx.x & 31;
    const int wave = threadIdx.x >> 5;
    const int m0 = blockIdx.y * 16;
    const int n0 = blockIdx.x * 128 + wave * 32;   // first of two 16-wide tiles
    if (n0 >= N || m0 >= M) return;                // wave-uniform exit
    const int rc = lane & 15;   // A: row within tile, B: col within tile
    const int kh = lane >> 4;   // K-half select
    const _Float16* Arow  = A  + (size_t)(m0 + rc) * lda;
    const _Float16* B0row = Bt + (size_t)(n0 + rc) * ldb;
    const _Float16* B1row = Bt + (size_t)(n0 + 16 + rc) * ldb;
    v8f acc0 = {}, acc1 = {};
#pragma unroll 2
    for (int k0 = 0; k0 < K; k0 += 32) {
        Frag16 fa, fb0, fb1;
        // A 16x32 f16: lanes0-15 K=0..15 ; lanes16-31 K=16..31
        fa.q[0] = *reinterpret_cast<const uint4*>(Arow + k0 + kh * 16);
        fa.q[1] = *reinterpret_cast<const uint4*>(Arow + k0 + kh * 16 + 8);
        // B 32x16 f16: lanes0-15 K=0..15 ; lanes16-31 K=16..31 (contiguous in Bt row)
        fb0.q[0] = *reinterpret_cast<const uint4*>(B0row + k0 + kh * 16);
        fb0.q[1] = *reinterpret_cast<const uint4*>(B0row + k0 + kh * 16 + 8);
        fb1.q[0] = *reinterpret_cast<const uint4*>(B1row + k0 + kh * 16);
        fb1.q[1] = *reinterpret_cast<const uint4*>(B1row + k0 + kh * 16 + 8);
        // stream-prefetch the B rows a few k-steps ahead (global_prefetch_b8)
        __builtin_prefetch(B0row + k0 + 256, 0, 1);
        __builtin_prefetch(B1row + k0 + 256, 0, 1);
        acc0 = wmma_f16(fa.v, fb0.v, acc0);
        acc1 = wmma_f16(fa.v, fb1.v, acc1);
    }
    // D layout: elem v -> row m0 + v + kh*8, col tile_n + (lane&15)
    const int cn0 = n0 + rc;
    const int cn1 = n0 + 16 + rc;
#pragma unroll
    for (int v = 0; v < 8; ++v) {
        const int cm = m0 + v + kh * 8;
        const float v0 = acc0[v];
        const float v1 = acc1[v];
        if (epi == EPI_TANH_ERR) {
            float e0 = Xres[(size_t)cm * ldx + cn0] - tanhf(v0);
            float e1 = Xres[(size_t)cm * ldx + cn1] - tanhf(v1);
            ErrF[(size_t)cm * N + cn0] = e0;
            ErrF[(size_t)cm * N + cn1] = e1;
            ErrH[(size_t)cm * N + cn0] = (_Float16)e0;
            ErrH[(size_t)cm * N + cn1] = (_Float16)e1;
        } else {
            Cout[(size_t)cm * ldc + cn0] = v0;
            Cout[(size_t)cm * ldc + cn1] = v1;
        }
    }
}

// ---------------- surprise gate + error EMA stats ----------------
__global__ __launch_bounds__(256)
void gate_stats(const float* __restrict__ err, float* __restrict__ e_mean,
                float* __restrict__ e_var, float* __restrict__ avg_S,
                float* __restrict__ S_out,
                const float* __restrict__ alpha_p, const float* __restrict__ gamma_p) {
    const int b = blockIdx.x, tid = threadIdx.x;
    float s_e2 = 0.f, s_m2 = 0.f, s_v = 0.f;
    for (int d = tid; d < kD; d += 256) {
        float e = err[b * kD + d];
        float m = e_mean[b * kD + d];
        float v = e_var[b * kD + d];
        s_e2 += e * e; s_m2 += m * m; s_v += v;
    }
    __shared__ float r0[256], r1[256], r2[256];
    r0[tid] = s_e2; r1[tid] = s_m2; r2[tid] = s_v;
    __syncthreads();
    for (int s = 128; s > 0; s >>= 1) {
        if (tid < s) { r0[tid] += r0[tid+s]; r1[tid] += r1[tid+s]; r2[tid] += r2[tid+s]; }
        __syncthreads();
    }
    if (tid == 0) {
        float err_norm = sqrtf(r0[0]);
        float baseline = sqrtf(r1[0]) + kEPS;
        float variance = r2[0] / (float)kD;
        float entropy = 0.5f * logf(kTWO_PI_E * (variance + kEPS));
        entropy = fminf(fmaxf(entropy, 0.f), 2.f);
        float tau_thr = 1.f + (*alpha_p) * entropy;
        float z = (err_norm / baseline - tau_thr) / ((*gamma_p) * 2.f);
        float S = 1.f / (1.f + expf(-z));
        S_out[b] = S;
        avg_S[b] = (1.f - kBETA_S) * avg_S[b] + kBETA_S * S;
    }
    __syncthreads();
    // EMA updates (S above used the pre-update stats)
    for (int d = tid; d < kD; d += 256) {
        float e = err[b * kD + d];
        float m = e_mean[b * kD + d];
        float v = e_var[b * kD + d];
        float mn = (1.f - kBETA) * m + kBETA * e;
        float dv = e - mn;
        float vn = (1.f - kBETA) * v + kBETA * dv * dv;
        e_mean[b * kD + d] = mn;
        e_var[b * kD + d] = vn;
    }
}

// m = mean(avg_S); also zero the per-b norm accumulators for this step
__global__ void reduce_m(const float* __restrict__ avg_S, float* __restrict__ m_out,
                         float* __restrict__ normU2, float* __restrict__ normT2) {
    __shared__ float r[kB];
    r[threadIdx.x] = avg_S[threadIdx.x];
    normU2[threadIdx.x] = 0.f;
    normT2[threadIdx.x] = 0.f;
    __syncthreads();
    for (int s = kB / 2; s > 0; s >>= 1) {
        if ((int)threadIdx.x < s) r[threadIdx.x] += r[threadIdx.x + s];
        __syncthreads();
    }
    if (threadIdx.x == 0) m_out[0] = r[0] / (float)kB;
}

// ---------------- Hebbian fast-weight update + norm (atomic-accumulated) ----
// grid (kB, kChunks) x 256 threads; float4 over the contiguous R axis
__global__ __launch_bounds__(256)
void U_update(float* __restrict__ U, const float* __restrict__ Utgt,
              const float* __restrict__ h32, const float* __restrict__ eV,
              const float* __restrict__ eta, const float* __restrict__ S,
              float* __restrict__ normU2) {
    const int b = blockIdx.x, tid = threadIdx.x;
    const float Sb = S[b];
    const size_t base = (size_t)b * kH * kR;
    const int per = (kH * kR) / kChunks;            // 16384
    const int start = blockIdx.y * per;
    float acc = 0.f;
    for (int i0 = start + tid * 4; i0 < start + per; i0 += 256 * 4) {
        const int hh = i0 >> 6;                     // kR == 64, same for all 4
        const int rr = i0 & 63;
        const float hs = (eta[hh] * Sb) * h32[b * kH + hh];
        float4 u4  = *reinterpret_cast<float4*>(U + base + i0);
        float4 ut4 = *reinterpret_cast<const float4*>(Utgt + base + i0);
        float4 ev4 = *reinterpret_cast<const float4*>(eV + b * kR + rr);
        float4 un;
        un.x = u4.x + (-kFORGET * (u4.x - ut4.x) + hs * ev4.x) * kDT;
        un.y = u4.y + (-kFORGET * (u4.y - ut4.y) + hs * ev4.y) * kDT;
        un.z = u4.z + (-kFORGET * (u4.z - ut4.z) + hs * ev4.z) * kDT;
        un.w = u4.w + (-kFORGET * (u4.w - ut4.w) + hs * ev4.w) * kDT;
        *reinterpret_cast<float4*>(U + base + i0) = un;
        acc += un.x * un.x + un.y * un.y + un.z * un.z + un.w * un.w;
    }
    __shared__ float red[256];
    red[tid] = acc;
    __syncthreads();
    for (int s = 128; s > 0; s >>= 1) {
        if (tid < s) red[tid] += red[tid + s];
        __syncthreads();
    }
    if (tid == 0) atomicAdd(&normU2[b], red[0]);
}

// ---------------- U rescale + fast-weight readout + LTC h update ----------------
__global__ __launch_bounds__(256)
void fuse_h(float* __restrict__ U, const float* __restrict__ normU2,
            const float* __restrict__ xV, const float* __restrict__ xB,
            const float* __restrict__ errW, const float* __restrict__ S,
            float* __restrict__ h32, _Float16* __restrict__ h16,
            float* __restrict__ out,
            const float* __restrict__ kappa_p, const float* __restrict__ tau_p, int t) {
    const int b = blockIdx.y;
    const int h = blockIdx.x * 256 + threadIdx.x;
    __shared__ float xv[kR];
    if (threadIdx.x < kR) xv[threadIdx.x] = xV[((size_t)b * kT + t) * kR + threadIdx.x];
    __syncthreads();
    const float scale = fminf(kTARGET / (sqrtf(normU2[b]) + kEPS), 2.f);
    float* Up = U + ((size_t)b * kH + h) * kR;
    float dot = 0.f;
#pragma unroll
    for (int r = 0; r < kR; r += 4) {
        float4 u4 = *reinterpret_cast<float4*>(Up + r);
        u4.x *= scale; u4.y *= scale; u4.z *= scale; u4.w *= scale;
        *reinterpret_cast<float4*>(Up + r) = u4;
        dot += u4.x * xv[r] + u4.y * xv[r+1] + u4.z * xv[r+2] + u4.w * xv[r+3];
    }
    const float Sb = S[b];
    float u_eff = (1.f + (*kappa_p) * Sb) * xB[((size_t)b * kT + t) * kH + h] + 0.1f * dot;
    float tau_l = fminf(fmaxf((*tau_p) / (1.f + Sb * kSURP), 0.01f), 50.f);
    float dg = fminf(fmaxf(kDT / (tau_l + kDT), 0.01f), 0.5f);
    float hold = h32[b * kH + h];
    float h_ltc = (1.f - dg) * hold + dg * tanhf(u_eff);
    float hn = (h_ltc + errW[b * kH + h]) * 0.99f + hold * 0.01f;
    h32[b * kH + h] = hn;
    h16[b * kH + h] = (_Float16)hn;
    out[((size_t)b * kT + t) * kH + h] = hn;
}

// ---------------- sleep consolidation of U_tgt ----------------
__global__ __launch_bounds__(256)
void Utgt_update(const float* __restrict__ U, float* __restrict__ Utgt,
                 const float* __restrict__ m_p, float* __restrict__ normT2) {
    const float m = m_p[0];
    if (m <= kSMIN) return;
    const int b = blockIdx.x, tid = threadIdx.x;
    const size_t base = (size_t)b * kH * kR;
    const int per = (kH * kR) / kChunks;
    const int start = blockIdx.y * per;
    float acc = 0.f;
    for (int i0 = start + tid * 4; i0 < start + per; i0 += 256 * 4) {
        float4 ut4 = *reinterpret_cast<float4*>(Utgt + base + i0);
        float4 u4  = *reinterpret_cast<const float4*>(U + base + i0);
        float4 c;
        c.x = ut4.x + kSLEEP * m * (u4.x - ut4.x);
        c.y = ut4.y + kSLEEP * m * (u4.y - ut4.y);
        c.z = ut4.z + kSLEEP * m * (u4.z - ut4.z);
        c.w = ut4.w + kSLEEP * m * (u4.w - ut4.w);
        *reinterpret_cast<float4*>(Utgt + base + i0) = c;
        acc += c.x * c.x + c.y * c.y + c.z * c.z + c.w * c.w;
    }
    __shared__ float red[256];
    red[tid] = acc;
    __syncthreads();
    for (int s = 128; s > 0; s >>= 1) {
        if (tid < s) red[tid] += red[tid + s];
        __syncthreads();
    }
    if (tid == 0) atomicAdd(&normT2[b], red[0]);
}

__global__ __launch_bounds__(256)
void Utgt_scale(float* __restrict__ Utgt, const float* __restrict__ normT2,
                const float* __restrict__ m_p) {
    if (m_p[0] <= kSMIN) return;
    const int b = blockIdx.y;
    const size_t i = ((size_t)blockIdx.x * 256 + threadIdx.x) * 4;
    const float sc = fminf(kTARGET / (sqrtf(normT2[b]) + kEPS), 2.f);
    float4* p = reinterpret_cast<float4*>(Utgt + (size_t)b * kH * kR + i);
    float4 v = *p;
    v.x *= sc; v.y *= sc; v.z *= sc; v.w *= sc;
    *p = v;
}

// ---------------- host ----------------
extern "C" void kernel_launch(void* const* d_in, const int* in_sizes, int n_in,
                              void* d_out, int out_size, void* d_ws, size_t ws_size,
                              hipStream_t stream) {
    const float* x_seq = (const float*)d_in[0];   // [B,T,D]
    const float* C     = (const float*)d_in[1];   // [H,D]
    const float* W     = (const float*)d_in[2];   // [D,H]
    const float* Bb    = (const float*)d_in[3];   // [D,H]
    const float* V     = (const float*)d_in[4];   // [D,R]
    const float* eta   = (const float*)d_in[5];   // [H]
    const float* alpha = (const float*)d_in[6];
    const float* gamma = (const float*)d_in[7];
    const float* kappa = (const float*)d_in[8];
    const float* tau   = (const float*)d_in[9];
    float* out = (float*)d_out;                    // [B,T,H]

    char* p = (char*)d_ws;
    auto carve = [&](size_t n) { char* r = p; p += (n + 255) & ~(size_t)255; return (void*)r; };
    _Float16* x16   = (_Float16*)carve((size_t)kB * kT * kD * 2);
    _Float16* Ct16  = (_Float16*)carve((size_t)kD * kH * 2);   // [D,H] = C^T
    _Float16* Wt16  = (_Float16*)carve((size_t)kH * kD * 2);   // [H,D] = W^T
    _Float16* Bt16  = (_Float16*)carve((size_t)kH * kD * 2);   // [H,D] = B_base^T
    _Float16* Vt16  = (_Float16*)carve((size_t)kR * kD * 2);   // [R,D] = V^T
    float* xB    = (float*)carve((size_t)kB * kT * kH * 4);    // x @ B_base
    float* xV    = (float*)carve((size_t)kB * kT * kR * 4);    // x @ V
    float* U     = (float*)carve((size_t)kB * kH * kR * 4);
    float* Utgt  = (float*)carve((size_t)kB * kH * kR * 4);
    float* h32   = (float*)carve((size_t)kB * kH * 4);
    _Float16* h16 = (_Float16*)carve((size_t)kB * kH * 2);
    float* err32 = (float*)carve((size_t)kB * kD * 4);
    _Float16* err16 = (_Float16*)carve((size_t)kB * kD * 2);
    float* errW  = (float*)carve((size_t)kB * kH * 4);
    float* eVb   = (float*)carve((size_t)kB * kR * 4);
    float* e_mean = (float*)carve((size_t)kB * kD * 4);
    float* e_var  = (float*)carve((size_t)kB * kD * 4);
    float* avg_S  = (float*)carve((size_t)kB * 4);
    float* Sbuf   = (float*)carve((size_t)kB * 4);
    float* normU2 = (float*)carve((size_t)kB * 4);
    float* normT2 = (float*)carve((size_t)kB * 4);
    float* m_sc   = (float*)carve(4);

    // --- prep: f16 conversions + transposed weights ---
    {
        int n = kB * kT * kD;
        cvt_f16<<<(n + 255) / 256, 256, 0, stream>>>(x_seq, x16, n);
        cvt_f16_T<<<(kH * kD + 255) / 256, 256, 0, stream>>>(C,  Ct16, kH, kD);
        cvt_f16_T<<<(kD * kH + 255) / 256, 256, 0, stream>>>(W,  Wt16, kD, kH);
        cvt_f16_T<<<(kD * kH + 255) / 256, 256, 0, stream>>>(Bb, Bt16, kD, kH);
        cvt_f16_T<<<(kD * kR + 255) / 256, 256, 0, stream>>>(V,  Vt16, kD, kR);
    }
    // --- hoisted input GEMMs: xB = x @ B_base, xV = x @ V (M = B*T = 4096) ---
    {
        dim3 g1(kH / 128, (kB * kT) / 16);
        gemm_f16<<<g1, 128, 0, stream>>>(x16, kD, Bt16, kD, kB * kT, kH, kD,
                                         xB, kH, EPI_STORE, nullptr, 0, nullptr, nullptr);
        dim3 g2((kR + 127) / 128, (kB * kT) / 16);
        gemm_f16<<<g2, 128, 0, stream>>>(x16, kD, Vt16, kD, kB * kT, kR, kD,
                                         xV, kR, EPI_STORE, nullptr, 0, nullptr, nullptr);
    }
    // --- zero-init scan state ---
    hipMemsetAsync(h32,  0, (size_t)kB * kH * 4, stream);
    hipMemsetAsync(h16,  0, (size_t)kB * kH * 2, stream);
    hipMemsetAsync(U,    0, (size_t)kB * kH * kR * 4, stream);
    hipMemsetAsync(Utgt, 0, (size_t)kB * kH * kR * 4, stream);
    hipMemsetAsync(e_mean, 0, (size_t)kB * kD * 4, stream);
    hipMemsetAsync(e_var,  0, (size_t)kB * kD * 4, stream);
    hipMemsetAsync(avg_S,  0, (size_t)kB * 4, stream);

    // --- sequential scan over T ---
    for (int t = 0; t < kT; ++t) {
        // 1) err = x_t - tanh(h @ C)   [B,D]  (WMMA, fused epilogue)
        {
            dim3 g(kD / 128, kB / 16);
            gemm_f16<<<g, 128, 0, stream>>>(h16, kH, Ct16, kH, kB, kD, kH,
                                            err32, kD, EPI_TANH_ERR,
                                            x_seq + (size_t)t * kD, kT * kD,
                                            err32, err16);
        }
        // 2) surprise gate + error EMA stats
        gate_stats<<<kB, 256, 0, stream>>>(err32, e_mean, e_var, avg_S, Sbuf, alpha, gamma);
        // 3) m = mean(avg_S); zero norm accumulators
        reduce_m<<<1, kB, 0, stream>>>(avg_S, m_sc, normU2, normT2);
        // 4) eV = err @ V   [B,R]
        {
            dim3 g((kR + 127) / 128, kB / 16);
            gemm_f16<<<g, 128, 0, stream>>>(err16, kD, Vt16, kD, kB, kR, kD,
                                            eVb, kR, EPI_STORE, nullptr, 0, nullptr, nullptr);
        }
        // 5) errW = err @ W   [B,H]
        {
            dim3 g(kH / 128, kB / 16);
            gemm_f16<<<g, 128, 0, stream>>>(err16, kD, Wt16, kD, kB, kH, kD,
                                            errW, kH, EPI_STORE, nullptr, 0, nullptr, nullptr);
        }
        // 6) U += dt * (-forget*(U-Utgt) + eta*S*h⊗eV), per-b norm (atomic)
        {
            dim3 g(kB, kChunks);
            U_update<<<g, 256, 0, stream>>>(U, Utgt, h32, eVb, eta, Sbuf, normU2);
        }
        // 7) rescale U, u_fast = U·(xV), LTC integrate h, write out[:,t,:]
        {
            dim3 g(kH / 256, kB);
            fuse_h<<<g, 256, 0, stream>>>(U, normU2, xV, xB, errW, Sbuf,
                                          h32, h16, out, kappa, tau, t);
        }
        // 8-9) sleep consolidation of U_tgt (uses scaled U) + rescale
        {
            dim3 g(kB, kChunks);
            Utgt_update<<<g, 256, 0, stream>>>(U, Utgt, m_sc, normT2);
        }
        {
            dim3 g((kH * kR) / (256 * 4), kB);
            Utgt_scale<<<g, 256, 0, stream>>>(Utgt, normT2, m_sc);
        }
    }
    (void)in_sizes; (void)n_in; (void)out_size; (void)ws_size;
}
